// SelfAttention_30030411334459
// MI455X (gfx1250) — matrix operand order repaired and checked by reading, not compiled, not run on hardware
//
#include <hip/hip_runtime.h>

typedef __bf16 bf16_t;
typedef __attribute__((ext_vector_type(16))) __bf16 v16bf;
typedef __attribute__((ext_vector_type(8)))  __bf16 v8bf;
typedef __attribute__((ext_vector_type(8)))  float  v8f;

#define NB     8
#define NH     8
#define HD     64
#define CDIM   512
#define SREAL  1025
#define SPAD   1040
#define MP     (NB*SPAD)      /* 8320 padded token rows */
#define NKT    (SPAD/16)      /* 65 key tiles */
#define ATT_SCALE 0.125f      /* 1/sqrt(64) */

__device__ __forceinline__ v16bf ld16(const bf16_t* p0, const bf16_t* p1){
  v8bf a = *(const v8bf*)p0;
  v8bf b = *(const v8bf*)p1;
  return __builtin_shufflevector(a, b, 0,1,2,3,4,5,6,7,8,9,10,11,12,13,14,15);
}
__device__ __forceinline__ v8f wmma_bf(v16bf a, v16bf b, v8f c){
  return __builtin_amdgcn_wmma_f32_16x16x32_bf16(false, a, false, b, (short)0, c, false, false);
}
__device__ __forceinline__ float gelu_exact(float x){
  return 0.5f*x*(1.0f + erff(x*0.70710678118654752f));
}

// ---------------- weight transpose f32(K,N) -> bf16(N,K), 512x512 ----------------
__global__ __launch_bounds__(256) void transpose_k(const float* __restrict__ in, bf16_t* __restrict__ out){
  int idx = blockIdx.x*256 + threadIdx.x;     // 0..262143
  int n = idx >> 9, k = idx & 511;
  out[idx] = (bf16_t)in[k*CDIM + n];
}

// ---------------- pack x -> seq f32 + bf16 (token rows 0..1023) ----------------
__global__ __launch_bounds__(256) void pack_x_k(const float* __restrict__ x, float* __restrict__ seq,
                                                bf16_t* __restrict__ seqb){
  size_t e0 = ((size_t)blockIdx.x*256 + threadIdx.x)*4;   // grid 4096 -> exactly 4194304
  #pragma unroll
  for (int i=0;i<4;i++){
    size_t e = e0 + i;
    int b = (int)(e >> 19);
    int r = (int)(e & 524287);
    int t = r >> 9, c = r & 511;
    size_t di = ((size_t)(b*SPAD + t) << 9) + c;
    float v = x[e];
    seq[di] = v; seqb[di] = (bf16_t)v;
  }
}

// ---------------- style embedding row (token 1024) + zero pad rows ----------------
__global__ __launch_bounds__(256) void style_k(const float* __restrict__ style, const float* __restrict__ emb_w,
                                               const float* __restrict__ emb_b,
                                               float* __restrict__ seq, bf16_t* __restrict__ seqb){
  int b = blockIdx.x, tid = threadIdx.x;
  for (int n = tid; n < CDIM; n += 256){
    float acc = emb_b[n];
    for (int m=0;m<CDIM;m++) acc += style[b*CDIM + m] * emb_w[m*CDIM + n];
    size_t di = ((size_t)(b*SPAD + 1024) << 9) + n;
    seq[di] = acc; seqb[di] = (bf16_t)acc;
  }
  for (int i = tid; i < 15*CDIM; i += 256){       // zero rows 1025..1039
    int t = 1025 + (i >> 9), n = i & 511;
    size_t di = ((size_t)(b*SPAD + t) << 9) + n;
    seq[di] = 0.0f; seqb[di] = (bf16_t)0.0f;
  }
}

// ---------------- per-head W_h[k][n] = scale * sum_m q_w[m,h,k]*v_w[m,h,n] ----------------
__global__ __launch_bounds__(256) void headw_k(const float* __restrict__ q_w, const float* __restrict__ v_w,
                                               float* __restrict__ W){
  int h = blockIdx.x, tid = threadIdx.x;
  for (int o = tid; o < 4096; o += 256){
    int k = o >> 6, n = o & 63;
    float acc = 0.f;
    for (int m=0;m<CDIM;m++) acc += q_w[m*CDIM + h*HD + k] * v_w[m*CDIM + h*HD + n];
    W[h*4096 + o] = acc * ATT_SCALE;
  }
}

// ---------------- generic bf16 WMMA GEMM: C(MxN) = A(MxK) * Bt(NxK)^T + bias ----------------
__global__ __launch_bounds__(256) void gemm_bf16_k(const bf16_t* __restrict__ A, const bf16_t* __restrict__ Bt,
                                                   const float* __restrict__ bias, float* __restrict__ C,
                                                   int M, int N, int K){
  int tid = threadIdx.x;
  int l = tid & 31, l15 = l & 15, lh = l >> 4, w = tid >> 5;
  int m0 = blockIdx.x*128 + (w & 3)*32;
  int n0 = blockIdx.y*64  + (w >> 2)*32;
  v8f c[2][2] = {};
  for (int kk = 0; kk < K; kk += 32){
    v16bf a[2], bb[2];
    #pragma unroll
    for (int i = 0; i < 2; i++){
      const bf16_t* p = A + (size_t)(m0 + i*16 + l15)*K + kk + lh*8;
      a[i] = ld16(p, p + 16);
    }
    #pragma unroll
    for (int j = 0; j < 2; j++){
      const bf16_t* p = Bt + (size_t)(n0 + j*16 + l15)*K + kk + lh*8;
      bb[j] = ld16(p, p + 16);
    }
    #pragma unroll
    for (int i=0;i<2;i++)
      #pragma unroll
      for (int j=0;j<2;j++)
        c[i][j] = wmma_bf(a[i], bb[j], c[i][j]);
  }
  // hoisted bias: one uniform select + at most 2 loads per lane
  float bv[2];
  #pragma unroll
  for (int j=0;j<2;j++) bv[j] = bias ? bias[n0 + j*16 + l15] : 0.0f;
  #pragma unroll
  for (int i=0;i<2;i++)
    #pragma unroll
    for (int j=0;j<2;j++){
      float* cp = C + (size_t)(m0 + i*16 + lh*8)*N + (n0 + j*16 + l15);
      #pragma unroll
      for (int r=0;r<8;r++){
        cp[0] = c[i][j][r] + bv[j];
        cp += N;
      }
    }
}

// ---------------- Q prep: Qbf(B,H,SPAD,64), sq = |q|^2, Vt(B,H,64,SPAD) = (q @ W_h)^T ----------------
__global__ __launch_bounds__(256) void qprep_k(const float* __restrict__ Q, const float* __restrict__ W,
                                               bf16_t* __restrict__ Qbf, float* __restrict__ sq,
                                               bf16_t* __restrict__ Vt){
  __shared__ float qs[16*64];
  __shared__ float ws[64*64];
  int bx = blockIdx.x;
  int sc = bx % NKT;
  int h  = (bx / NKT) & 7;
  int b  = bx / (NKT*NH);
  int s0 = sc*16, tid = threadIdx.x;
  for (int i = tid; i < 4096; i += 256) ws[i] = W[h*4096 + i];
  for (int i = tid; i < 1024; i += 256){
    int t = i >> 6, k = i & 63;
    float v = Q[(size_t)(b*SPAD + s0 + t)*CDIM + h*HD + k];
    qs[i] = v;
    Qbf[(size_t)((b*NH + h)*SPAD + s0 + t)*HD + k] = (bf16_t)v;
  }
  __syncthreads();
  if (tid < 16){
    float acc = 0.f;
    const float* qr0 = qs + tid*64;
    for (int k=0;k<64;k++) acc += qr0[k]*qr0[k];
    sq[(size_t)(b*NH + h)*SPAD + s0 + tid] = acc;
  }
  int t = tid >> 4, n0 = (tid & 15)*4;
  float a0=0.f,a1=0.f,a2=0.f,a3=0.f;
  const float* qr = qs + t*64;
  for (int k=0;k<64;k++){
    float qv = qr[k];
    const float* wr = ws + k*64 + n0;
    a0 += qv*wr[0]; a1 += qv*wr[1]; a2 += qv*wr[2]; a3 += qv*wr[3];
  }
  size_t base = (size_t)(b*NH + h)*HD*SPAD;
  int s = s0 + t;
  Vt[base + (size_t)(n0+0)*SPAD + s] = (bf16_t)a0;
  Vt[base + (size_t)(n0+1)*SPAD + s] = (bf16_t)a1;
  Vt[base + (size_t)(n0+2)*SPAD + s] = (bf16_t)a2;
  Vt[base + (size_t)(n0+3)*SPAD + s] = (bf16_t)a3;
}

// ---------------- attention: per (b,h,16-query tile) two-pass softmax, WMMA scores and P*V ----------------
__global__ __launch_bounds__(256) void attn_k(const bf16_t* __restrict__ Qbf, const float* __restrict__ sq,
                                              const bf16_t* __restrict__ Vt, bf16_t* __restrict__ O){
  constexpr int LROW = 1044;   // f32 logit row stride (bank-conflict pad)
  constexpr int PROW = 1048;   // bf16 P row stride (16B aligned rows)
  extern __shared__ char smem[];
  float*  Ls   = (float*)smem;                                   // [16][LROW] f32
  bf16_t* Ps   = (bf16_t*)(smem + 16*LROW*4);                    // [16][PROW] bf16
  float*  sqt  = (float*)(smem + 16*LROW*4 + 16*PROW*2);         // [16]
  float*  rinv = sqt + 16;                                       // [16]
  float*  Op   = rinv + 16;                                      // [8][16][64] f32 partials

  int bx = blockIdx.x;
  int qt = bx % NKT;
  int h  = (bx / NKT) & 7;
  int b  = bx / (NKT*NH);
  int t0 = qt*16;
  int tid = threadIdx.x, l = tid & 31, l15 = l & 15, lh = l >> 4, w = tid >> 5;

  const bf16_t* Qbh = Qbf + (size_t)(b*NH + h)*SPAD*HD;
  const float*  sqb = sq  + (size_t)(b*NH + h)*SPAD;
  const bf16_t* Vtb = Vt  + (size_t)(b*NH + h)*HD*SPAD;

  if (tid < 16) sqt[tid] = sqb[t0 + tid];
  __syncthreads();

  float sqtr[8];
  #pragma unroll
  for (int r=0;r<8;r++) sqtr[r] = sqt[r + lh*8];

  v16bf aq0, aq1;
  {
    const bf16_t* qr = Qbh + (size_t)(t0 + l15)*HD + lh*8;
    aq0 = ld16(qr,      qr + 16);
    aq1 = ld16(qr + 32, qr + 48);
  }

  // phase 1: logits[t, s] = (2*q_t.q_s - |q_t|^2 - |q_s|^2) * scale
  for (int kt = w; kt < NKT; kt += 8){
    int s0 = kt*16;
    int scol = s0 + l15;
    const bf16_t* kr = Qbh + (size_t)scol*HD + lh*8;
    v16bf b0 = ld16(kr,      kr + 16);
    v16bf b1 = ld16(kr + 32, kr + 48);
    v8f cc = {};
    cc = wmma_bf(aq0, b0, cc);
    cc = wmma_bf(aq1, b1, cc);
    float sqs = sqb[scol];
    #pragma unroll
    for (int r=0;r<8;r++){
      int m = r + lh*8;
      float val = (2.0f*cc[r] - sqtr[r] - sqs) * ATT_SCALE;
      if (scol >= SREAL) val = -1.0e30f;
      Ls[m*LROW + scol] = val;
    }
  }
  __syncthreads();

  // phase 2: row softmax (unnormalized P in bf16, keep 1/rowsum)
  {
    int row = tid >> 4, sub = tid & 15;
    float* lr = Ls + row*LROW;
    float mx = -3.0e38f;
    for (int s2 = sub; s2 < SPAD; s2 += 16) mx = fmaxf(mx, lr[s2]);
    #pragma unroll
    for (int o=1;o<16;o<<=1) mx = fmaxf(mx, __shfl_xor(mx, o, 32));
    float sum = 0.f;
    bf16_t* pr = Ps + row*PROW;
    for (int s2 = sub; s2 < SPAD; s2 += 16){
      float p = __expf(lr[s2] - mx);
      sum += p;
      pr[s2] = (bf16_t)p;
    }
    #pragma unroll
    for (int o=1;o<16;o<<=1) sum += __shfl_xor(sum, o, 32);
    if (sub == 0) rinv[row] = 1.0f / sum;
  }
  __syncthreads();

  // phase 3: O(16x64) = P(16xSPAD) * V(SPADx64), waves split K, reduce in LDS
  v8f oc[4] = {};
  for (int st = w; st < NKT; st += 8){
    int s0 = st*16;
    const bf16_t* pr = Ps + (size_t)l15*PROW + s0 + lh*8;
    v16bf ap = ld16(pr, pr + 16);
    #pragma unroll
    for (int q=0;q<4;q++){
      const bf16_t* vr = Vtb + (size_t)(q*16 + l15)*SPAD + s0 + lh*8;
      v16bf bv = ld16(vr, vr + 16);
      oc[q] = wmma_bf(ap, bv, oc[q]);
    }
  }
  #pragma unroll
  for (int q=0;q<4;q++)
    #pragma unroll
    for (int r=0;r<8;r++){
      int m = r + lh*8, n = q*16 + l15;
      Op[(w*16 + m)*64 + n] = oc[q][r];
    }
  __syncthreads();
  for (int o = tid; o < 1024; o += 256){
    int m = o >> 6, n = o & 63;
    float acc = 0.f;
    #pragma unroll
    for (int ww=0;ww<8;ww++) acc += Op[(ww*16 + m)*64 + n];
    acc *= rinv[m];
    int t = t0 + m;
    if (t < SREAL) O[(size_t)(b*SPAD + t)*CDIM + h*HD + n] = (bf16_t)acc;
  }
}

// ---------------- LN1: out1 = LN(attn_out + seq), also bf16 gelu(out1) ----------------
__global__ __launch_bounds__(256) void ln1_k(const float* __restrict__ xin, const float* __restrict__ res,
                                             const float* __restrict__ gw, const float* __restrict__ bw,
                                             float* __restrict__ out1, bf16_t* __restrict__ gb){
  __shared__ float red[16];
  size_t row = blockIdx.x;
  int tid = threadIdx.x, l = tid & 31, w = tid >> 5;
  const float* xr = xin + row*CDIM;
  const float* rr = res + row*CDIM;
  float v0 = xr[tid] + rr[tid];
  float v1 = xr[tid+256] + rr[tid+256];
  float s = v0 + v1, s2 = v0*v0 + v1*v1;
  #pragma unroll
  for (int o=16;o>0;o>>=1){ s += __shfl_xor(s,o,32); s2 += __shfl_xor(s2,o,32); }
  if (l == 0){ red[w] = s; red[8+w] = s2; }
  __syncthreads();
  float ts=0.f, ts2=0.f;
  #pragma unroll
  for (int i=0;i<8;i++){ ts += red[i]; ts2 += red[8+i]; }
  float mu = ts*(1.0f/512.0f);
  float ri = rsqrtf(ts2*(1.0f/512.0f) - mu*mu + 1e-5f);
  float y0 = (v0-mu)*ri*gw[tid] + bw[tid];
  float y1 = (v1-mu)*ri*gw[tid+256] + bw[tid+256];
  out1[row*CDIM + tid] = y0;
  out1[row*CDIM + tid + 256] = y1;
  gb[row*CDIM + tid] = (bf16_t)gelu_exact(y0);
  gb[row*CDIM + tid + 256] = (bf16_t)gelu_exact(y1);
}

// ---------------- elementwise gelu f32 -> bf16 ----------------
__global__ __launch_bounds__(256) void gelu_k(const float* __restrict__ in, bf16_t* __restrict__ out, int n){
  int i0 = (blockIdx.x*256 + threadIdx.x)*4;
  #pragma unroll
  for (int i=0;i<4;i++){
    int ii = i0 + i;
    if (ii < n) out[ii] = (bf16_t)gelu_exact(in[ii]);
  }
}

// ---------------- LN2 + NCHW store (only tokens 0..1023) ----------------
__global__ __launch_bounds__(256) void ln2_k(const float* __restrict__ xin, const float* __restrict__ res,
                                             const float* __restrict__ gw, const float* __restrict__ bw,
                                             float* __restrict__ out){
  __shared__ float red[16];
  int bx = blockIdx.x;
  int b = bx >> 10, t = bx & 1023;
  size_t row = (size_t)b*SPAD + t;
  int tid = threadIdx.x, l = tid & 31, w = tid >> 5;
  const float* xr = xin + row*CDIM;
  const float* rr = res + row*CDIM;
  float v0 = xr[tid] + rr[tid];
  float v1 = xr[tid+256] + rr[tid+256];
  float s = v0 + v1, s2 = v0*v0 + v1*v1;
  #pragma unroll
  for (int o=16;o>0;o>>=1){ s += __shfl_xor(s,o,32); s2 += __shfl_xor(s2,o,32); }
  if (l == 0){ red[w] = s; red[8+w] = s2; }
  __syncthreads();
  float ts=0.f, ts2=0.f;
  #pragma unroll
  for (int i=0;i<8;i++){ ts += red[i]; ts2 += red[8+i]; }
  float mu = ts*(1.0f/512.0f);
  float ri = rsqrtf(ts2*(1.0f/512.0f) - mu*mu + 1e-5f);
  float y0 = (v0-mu)*ri*gw[tid] + bw[tid];
  float y1 = (v1-mu)*ri*gw[tid+256] + bw[tid+256];
  out[((size_t)b*CDIM + tid)*1024 + t] = y0;
  out[((size_t)b*CDIM + tid + 256)*1024 + t] = y1;
}

extern "C" void kernel_launch(void* const* d_in, const int* in_sizes, int n_in,
                              void* d_out, int out_size, void* d_ws, size_t ws_size,
                              hipStream_t stream)
{
  (void)in_sizes; (void)n_in; (void)out_size; (void)ws_size;
  const float* x     = (const float*)d_in[0];
  const float* style = (const float*)d_in[1];
  const float* emb_w = (const float*)d_in[2];
  const float* emb_b = (const float*)d_in[3];
  const float* q_w   = (const float*)d_in[4];
  const float* v_w   = (const float*)d_in[5];
  const float* out_w = (const float*)d_in[6];
  const float* out_b = (const float*)d_in[7];
  const float* ff_w1 = (const float*)d_in[8];
  const float* ff_b1 = (const float*)d_in[9];
  const float* ff_w2 = (const float*)d_in[10];
  const float* ff_b2 = (const float*)d_in[11];
  const float* ln1g  = (const float*)d_in[12];
  const float* ln1b  = (const float*)d_in[13];
  const float* ln2g  = (const float*)d_in[14];
  const float* ln2b  = (const float*)d_in[15];

  char* ws = (char*)d_ws;
  size_t off = 0;
  auto carve = [&](size_t bytes)->char*{
    char* p = ws + off;
    off += (bytes + 255) & ~(size_t)255;
    return p;
  };
  float*  seqF  = (float*) carve((size_t)MP*CDIM*4);   // seq f32 (residual 1)
  float*  bufA  = (float*) carve((size_t)MP*CDIM*4);   // Q f32 -> attn_out -> hdn -> out2
  float*  out1F = (float*) carve((size_t)MP*CDIM*4);   // out1 f32 (residual 2)
  bf16_t* bufB  = (bf16_t*)carve((size_t)MP*CDIM*2);   // seq_bf -> PXAV_bf
  bf16_t* bufC  = (bf16_t*)carve((size_t)MP*CDIM*2);   // Qbf -> gelu1_bf
  bf16_t* bufD  = (bf16_t*)carve((size_t)MP*CDIM*2);   // Vt_bf -> gelu2_bf
  bf16_t* wT0   = (bf16_t*)carve((size_t)CDIM*CDIM*2); // q_w^T
  bf16_t* wT1   = (bf16_t*)carve((size_t)CDIM*CDIM*2); // out_w^T
  bf16_t* wT2   = (bf16_t*)carve((size_t)CDIM*CDIM*2); // ff_w1^T
  bf16_t* wT3   = (bf16_t*)carve((size_t)CDIM*CDIM*2); // ff_w2^T
  float*  Wh    = (float*) carve((size_t)NH*HD*HD*4);  // per-head scale*q_w^T v_w
  float*  sqB   = (float*) carve((size_t)NB*NH*SPAD*4);

  transpose_k<<<1024,256,0,stream>>>(q_w,  wT0);
  transpose_k<<<1024,256,0,stream>>>(out_w,wT1);
  transpose_k<<<1024,256,0,stream>>>(ff_w1,wT2);
  transpose_k<<<1024,256,0,stream>>>(ff_w2,wT3);
  pack_x_k<<<4096,256,0,stream>>>(x, seqF, bufB);
  style_k<<<NB,256,0,stream>>>(style, emb_w, emb_b, seqF, bufB);
  headw_k<<<NH,256,0,stream>>>(q_w, v_w, Wh);

  dim3 gg(MP/128, CDIM/64);
  gemm_bf16_k<<<gg,256,0,stream>>>(bufB, wT0, nullptr, bufA, MP, CDIM, CDIM);  // Q = seq @ q_w
  qprep_k<<<NB*NH*NKT,256,0,stream>>>(bufA, Wh, bufC, sqB, bufD);

  size_t shmem = (size_t)16*1044*4 + (size_t)16*1048*2 + 64 + 64 + (size_t)8*16*64*4;
  (void)hipFuncSetAttribute((const void*)attn_k, hipFuncAttributeMaxDynamicSharedMemorySize, (int)shmem);
  attn_k<<<NB*NH*NKT,256,shmem,stream>>>(bufC, sqB, bufD, bufB);               // PXAV (bf16)

  gemm_bf16_k<<<gg,256,0,stream>>>(bufB, wT1, out_b, bufA, MP, CDIM, CDIM);    // attn_out
  ln1_k<<<MP,256,0,stream>>>(bufA, seqF, ln1g, ln1b, out1F, bufC);             // out1 + gelu(out1)
  gemm_bf16_k<<<gg,256,0,stream>>>(bufC, wT2, ff_b1, bufA, MP, CDIM, CDIM);    // hdn
  gelu_k<<<4160,256,0,stream>>>(bufA, bufD, MP*CDIM);                          // gelu(hdn)
  gemm_bf16_k<<<gg,256,0,stream>>>(bufD, wT3, ff_b2, bufA, MP, CDIM, CDIM);    // out2
  ln2_k<<<NB*1024,256,0,stream>>>(bufA, out1F, ln2g, ln2b, (float*)d_out);     // LN + NCHW
}
